// Functional_11123965296846
// MI455X (gfx1250) — matrix operand order repaired and checked by reading, not compiled, not run on hardware
//
#include <hip/hip_runtime.h>
#include <hip/hip_bf16.h>

typedef _Float16 h8   __attribute__((ext_vector_type(8)));
typedef _Float16 v16h __attribute__((ext_vector_type(16)));
typedef float    v8f  __attribute__((ext_vector_type(8)));
typedef int      vsi4 __attribute__((vector_size(16)));   // matches builtin param type

#define NZ      65536
#define BATCH   64
#define CHUNK   2048
#define NCHUNK  (NZ / CHUNK)          // 32
#define RE      49
#define ETAPS   (2*RE + 1)            // 99
#define RO      128
#define OTAPS   (2*RO + 1)            // 257
#define DZF     0.1f
#define RCF     5.0f
#define PI_F    3.14159265358979323846f

#define NWAVES   4
#define NTHREADS (NWAVES * 32)

#define SCH_STR 32                    // halves per scalar-row (64B, 16B aligned)
#define X1_STR  72                    // halves per x-row (144B, 16B aligned, conflict-friendly)

// workspace layout in floats
#define WS_ET   0                     // ETAPS*4 = 396
#define WS_OT   512                   // OTAPS*4 = 1028
#define WS_PART 2048                  // BATCH*NCHUNK = 2048

#define AS1 __attribute__((address_space(1)))
#define AS3 __attribute__((address_space(3)))

#if __has_builtin(__builtin_amdgcn_global_load_async_to_lds_b128)
#define HAVE_ASYNC_LDS 1
#endif

// branchless softplus: max(x,0) + log1p(exp(-|x|)).
// log argument is in [1,2] -> use raw v_log_f32 (log2) * ln2, no denorm guards.
__device__ __forceinline__ float softplusf(float x) {
    float e = __expf(-fabsf(x));
#if __has_builtin(__builtin_amdgcn_logf)
    return fmaxf(x, 0.0f) + 0.6931471805599453f * __builtin_amdgcn_logf(1.0f + e);
#else
    return fmaxf(x, 0.0f) + __logf(1.0f + e);
#endif
}

__device__ __forceinline__ void async_g2l_b128(void* lds, const void* g) {
#ifdef HAVE_ASYNC_LDS
    __builtin_amdgcn_global_load_async_to_lds_b128(
        (AS1 vsi4*)(g), (AS3 vsi4*)(lds), 0, 0);
#else
    *(float4*)lds = *(const float4*)g;
#endif
}

__device__ __forceinline__ void async_wait0() {
#ifdef HAVE_ASYNC_LDS
#if __has_builtin(__builtin_amdgcn_s_wait_asynccnt)
    __builtin_amdgcn_s_wait_asynccnt(0);
#else
    asm volatile("s_wait_asynccnt 0x0" ::: "memory");
#endif
#endif
}

// A-fragment (16x32 f16): lane row m, halves: K = 8*half+{0..7} and 16+8*half+{0..7}
__device__ __forceinline__ v16h ldfragA(const _Float16* base, int row, int rstr,
                                        int kbase, int half) {
    const _Float16* p = base + row * rstr + kbase + 8 * half;
    union { v16h v; h8 h[2]; } u;
    u.h[0] = *(const h8*)p;
    u.h[1] = *(const h8*)(p + 16);
    return u.v;
}

// B-fragment (32x16 f16): lane col n, K = 8*half+{0..7} and 16+8*half+{0..7}
__device__ __forceinline__ v16h ldfragB(const _Float16* base, int row, int rstr,
                                        int kbase, int half) {
    const _Float16* p = base + row * rstr + kbase + 8 * half;
    union { v16h v; h8 h[2]; } u;
    u.h[0] = *(const h8*)p;
    u.h[1] = *(const h8*)(p + 16);
    return u.v;
}

#define WMMA16(A, B, C) \
    __builtin_amdgcn_wmma_f32_16x16x32_f16(false, (A), false, (B), (short)0, (C), false, false)

// ---------------------------------------------------------------------------
// Kernel 1: build real-space filter taps from the w-MLP.
// Even channels: 99-tap normalized kernel. Odd channels: kernel convolved with
// truncated discrete spectral-derivative D[m] = (-1)^m (pi/L) / tan(pi m / N).
// ---------------------------------------------------------------------------
__global__ __launch_bounds__(128) void build_filters(
    const float* __restrict__ wW0, const float* __restrict__ wB0,
    const float* __restrict__ wW1, const float* __restrict__ wB1,
    const float* __restrict__ wW2, const float* __restrict__ wB2,
    float* __restrict__ ws)
{
    __shared__ float wr[8][ETAPS];
    __shared__ float csum[8];
    const int tid = threadIdx.x;

    for (int j = tid; j < ETAPS; j += 128) {
        float z = (float)(j - RE) * DZF;
        float u = (z / RCF) * (z / RCF);
        u = fminf(u, 1.0f);
        float pre = (1.0f - u) * (1.0f - u);
        float h1[64], h2[64];
        for (int o = 0; o < 64; ++o) h1[o] = softplusf(wW0[o] * u + wB0[o]);
        for (int o = 0; o < 64; ++o) {
            float a = wB1[o];
            for (int i = 0; i < 64; ++i) a += wW1[o * 64 + i] * h1[i];
            h2[o] = softplusf(a);
        }
        for (int c = 0; c < 8; ++c) {
            float a = wB2[c];
            for (int i = 0; i < 64; ++i) a += wW2[c * 64 + i] * h2[i];
            wr[c][j] = pre / (1.0f + a * a);
        }
    }
    __syncthreads();
    if (tid < 8) {
        float s = 0.0f;
        for (int j = 0; j < ETAPS; ++j) s += wr[tid][j];
        csum[tid] = s;
    }
    __syncthreads();
    for (int j = tid; j < 8 * ETAPS; j += 128) {
        int c = j / ETAPS, t = j % ETAPS;
        wr[c][t] /= csum[c];
    }
    __syncthreads();
    // even taps, interleaved by channel
    for (int j = tid; j < ETAPS; j += 128)
        for (int c = 0; c < 4; ++c)
            ws[WS_ET + j * 4 + c] = wr[c][j];
    // odd taps: O[c][t] = sum_s h[4+c][s] * D[t - s]
    const float L = DZF * (float)NZ;
    for (int j = tid; j < OTAPS; j += 128) {
        int t = j - RO;
        float a0 = 0.f, a1 = 0.f, a2 = 0.f, a3 = 0.f;
        for (int s = -RE; s <= RE; ++s) {
            int m = t - s;
            float d = 0.0f;
            if (m != 0) {
                float sgn = (m & 1) ? -1.0f : 1.0f;
                d = sgn * (PI_F / L) / tanf(PI_F * (float)m / (float)NZ);
            }
            a0 += wr[4 + 0][s + RE] * d;
            a1 += wr[4 + 1][s + RE] * d;
            a2 += wr[4 + 2][s + RE] * d;
            a3 += wr[4 + 3][s + RE] * d;
        }
        ws[WS_OT + j * 4 + 0] = a0;
        ws[WS_OT + j * 4 + 1] = a1;
        ws[WS_OT + j * 4 + 2] = a2;
        ws[WS_OT + j * 4 + 3] = a3;
    }
}

// ---------------------------------------------------------------------------
// Kernel 2: fused conv (8 channels) + pair products + f-MLP via WMMA + reduce.
// Grid: (NCHUNK, BATCH), 128 threads = 4 wave32.
// ---------------------------------------------------------------------------
__global__ __launch_bounds__(NTHREADS) void fused_main(
    const float* __restrict__ n_in,
    const float* __restrict__ F0, const float* __restrict__ Fb0,
    const float* __restrict__ F1, const float* __restrict__ Fb1,
    const float* __restrict__ F2, const float* __restrict__ Fb2,
    const float* __restrict__ ws_taps, float* __restrict__ ws_part)
{
    __shared__ __align__(16) float    nlds[CHUNK + 2 * RO];      // 9216 B
    __shared__ __align__(16) float    et[ETAPS * 4];             // 1584 B
    __shared__ __align__(16) float    ot[OTAPS * 4];             // 4112 B
    __shared__ __align__(16) _Float16 w1p[64 * 32];              // 4096 B (K=14 col = bias)
    __shared__ __align__(16) _Float16 w2p[64 * 64];              // 8192 B
    __shared__ __align__(16) _Float16 w3p[16 * 64];              // 2048 B (rows 14,15 zero)
    __shared__ float    b2s[64];
    __shared__ float    b3s[16];
    __shared__ __align__(16) _Float16 sch[NWAVES][32 * SCH_STR]; // 8192 B
    __shared__ __align__(16) _Float16 x1b[NWAVES][16 * X1_STR];  // 9216 B
    __shared__ float    red[NTHREADS];

    const int tid  = threadIdx.x;
    const int wv   = tid >> 5;
    const int lane = tid & 31;
    const int half = lane >> 4;
    const int nn   = lane & 15;
    const int b    = blockIdx.y;
    const int z0   = blockIdx.x * CHUNK;

    // circular n chunk with halo, staged via async global->LDS DMA (ASYNCcnt)
    const float* nrow = n_in + (size_t)b * NZ;
    {
        const int npieces = (CHUNK + 2 * RO) / 4;                // 16B pieces, wrap-safe
        for (int i = tid; i < npieces; i += NTHREADS) {
            int idx = (z0 - RO + i * 4 + NZ) & (NZ - 1);
            async_g2l_b128(&nlds[i * 4], &nrow[idx]);
            __builtin_prefetch(&nrow[(idx + CHUNK) & (NZ - 1)], 0, 1);
        }
    }

    for (int i = tid; i < ETAPS * 4; i += NTHREADS) et[i] = ws_taps[WS_ET + i];
    for (int i = tid; i < OTAPS * 4; i += NTHREADS) ot[i] = ws_taps[WS_OT + i];
    // f16 weights; fold layer-1 bias into K=14 column
    for (int i = tid; i < 64 * 32; i += NTHREADS) {
        int m = i >> 5, k = i & 31;
        float v = (k < 14) ? F0[m * 14 + k] : ((k == 14) ? Fb0[m] : 0.0f);
        w1p[i] = (_Float16)v;
    }
    for (int i = tid; i < 64 * 64; i += NTHREADS) w2p[i] = (_Float16)F1[i];
    for (int i = tid; i < 16 * 64; i += NTHREADS) {
        int m = i >> 6, k = i & 63;
        w3p[i] = (_Float16)((m < 14) ? F2[m * 64 + k] : 0.0f);
    }
    for (int i = tid; i < 64; i += NTHREADS) b2s[i] = Fb1[i];
    if (tid < 16) b3s[tid] = (tid < 14) ? Fb2[tid] : 0.0f;

    // zero scalar buffer, then set constant-1 bias channel at K=14
    for (int i = lane; i < 32 * SCH_STR; i += 32) sch[wv][i] = (_Float16)0.0f;
    for (int r = lane; r < 32; r += 32) sch[wv][r * SCH_STR + 14] = (_Float16)1.0f;

    async_wait0();          // wave's async DMA complete
    __syncthreads();        // visible block-wide

    float acc = 0.0f;
    for (int it = 0; it < CHUNK / (NWAVES * 32); ++it) {   // 16 iterations
        const int p = (it * NWAVES + wv) * 32 + lane;      // this lane's point in chunk

        // ---- 8-channel convolution for point p ----
        float e0 = 0.f, e1 = 0.f, e2 = 0.f, e3 = 0.f;
        {
            const float* xp = &nlds[p + (RO - RE)];
            for (int j = 0; j < ETAPS; ++j) {
                float x = xp[j];
                const float* t4 = &et[j * 4];
                e0 += t4[0] * x; e1 += t4[1] * x; e2 += t4[2] * x; e3 += t4[3] * x;
            }
        }
        float o0 = 0.f, o1 = 0.f, o2 = 0.f, o3 = 0.f;
        {
            const float* yp = &nlds[p + 2 * RO];
            for (int j = 0; j < OTAPS; ++j) {
                float x = yp[-j];                          // n[x - t], t = j - RO
                const float* t4 = &ot[j * 4];
                o0 += t4[0] * x; o1 += t4[1] * x; o2 += t4[2] * x; o3 += t4[3] * x;
            }
        }
        // ---- 14 scalars: 4 even + 10 odd pair products (i<=j) ----
        float sc[14];
        sc[0] = e0; sc[1] = e1; sc[2] = e2; sc[3] = e3;
        float od[4] = {o0, o1, o2, o3};
        int q = 4;
#pragma unroll
        for (int i = 0; i < 4; ++i)
#pragma unroll
            for (int j = i; j < 4; ++j)
                sc[q++] = od[i] * od[j];
#pragma unroll
        for (int c = 0; c < 14; ++c)
            sch[wv][lane * SCH_STR + c] = (_Float16)sc[c];

        // ---- MLP on two 16-point tiles via WMMA ----
        for (int tile = 0; tile < 2; ++tile) {
            const int prow = nn + 16 * tile;               // point row in sch

            // layer 1: [64x32] x [32x16]
            v16h bf = ldfragB(&sch[wv][0], prow, SCH_STR, 0, half);
            v8f  c1[4];
#pragma unroll
            for (int t = 0; t < 4; ++t) {
                v16h a = ldfragA(w1p, 16 * t + nn, 32, 0, half);
                v8f  z = {0.f, 0.f, 0.f, 0.f, 0.f, 0.f, 0.f, 0.f};
                c1[t] = WMMA16(a, bf, z);
            }
#pragma unroll
            for (int t = 0; t < 4; ++t) {                  // softplus -> x1
                h8 hx;
#pragma unroll
                for (int r = 0; r < 8; ++r) hx[r] = (_Float16)softplusf(c1[t][r]);
                *(h8*)&x1b[wv][nn * X1_STR + 16 * t + 8 * half] = hx;
            }

            // layer 2: [64x64] x [64x16]
            v16h b20 = ldfragB(&x1b[wv][0], nn, X1_STR, 0, half);
            v16h b21 = ldfragB(&x1b[wv][0], nn, X1_STR, 32, half);
            v8f  c2[4];
#pragma unroll
            for (int t = 0; t < 4; ++t) {
                v8f z = {0.f, 0.f, 0.f, 0.f, 0.f, 0.f, 0.f, 0.f};
                v16h a0 = ldfragA(w2p, 16 * t + nn, 64, 0, half);
                z = WMMA16(a0, b20, z);
                v16h a1 = ldfragA(w2p, 16 * t + nn, 64, 32, half);
                z = WMMA16(a1, b21, z);
                c2[t] = z;
            }
#pragma unroll
            for (int t = 0; t < 4; ++t) {                  // bias + softplus -> x2
                h8 hx;
#pragma unroll
                for (int r = 0; r < 8; ++r)
                    hx[r] = (_Float16)softplusf(c2[t][r] + b2s[16 * t + 8 * half + r]);
                *(h8*)&x1b[wv][nn * X1_STR + 16 * t + 8 * half] = hx;
            }

            // layer 3: [16x64] x [64x16]  (rows 14,15 of W3 are zero)
            v16h b30 = ldfragB(&x1b[wv][0], nn, X1_STR, 0, half);
            v16h b31 = ldfragB(&x1b[wv][0], nn, X1_STR, 32, half);
            v8f  c3 = {0.f, 0.f, 0.f, 0.f, 0.f, 0.f, 0.f, 0.f};
            v16h a30 = ldfragA(w3p, nn, 64, 0, half);
            c3 = WMMA16(a30, b30, c3);
            v16h a31 = ldfragA(w3p, nn, 64, 32, half);
            c3 = WMMA16(a31, b31, c3);

            // dot(scalars, f_val): channel m = 8*half + r.  s[14]=1 pairs with y=0,
            // s[15]=0, so padded channels contribute exactly zero.
            const _Float16* sp = &sch[wv][prow * SCH_STR + 8 * half];
#pragma unroll
            for (int r = 0; r < 8; ++r)
                acc += (c3[r] + b3s[8 * half + r]) * (float)sp[r];
        }
    }

    // deterministic block reduction
    red[tid] = acc;
    __syncthreads();
    for (int s = NTHREADS / 2; s > 0; s >>= 1) {
        if (tid < s) red[tid] += red[tid + s];
        __syncthreads();
    }
    if (tid == 0)
        ws_part[(size_t)b * NCHUNK + blockIdx.x] = red[0] * DZF;
}

// ---------------------------------------------------------------------------
// Kernel 3: deterministic final reduction per batch row (also inits d_out).
// ---------------------------------------------------------------------------
__global__ __launch_bounds__(64) void finalize(const float* __restrict__ ws_part,
                                               float* __restrict__ out)
{
    int bb = threadIdx.x;
    if (bb < BATCH) {
        float s = 0.0f;
        for (int i = 0; i < NCHUNK; ++i) s += ws_part[(size_t)bb * NCHUNK + i];
        out[bb] = s;
    }
}

extern "C" void kernel_launch(void* const* d_in, const int* in_sizes, int n_in,
                              void* d_out, int out_size, void* d_ws, size_t ws_size,
                              hipStream_t stream) {
    (void)in_sizes; (void)n_in; (void)out_size; (void)ws_size;
    const float* n   = (const float*)d_in[0];
    const float* wW0 = (const float*)d_in[1];
    const float* wB0 = (const float*)d_in[2];
    const float* wW1 = (const float*)d_in[3];
    const float* wB1 = (const float*)d_in[4];
    const float* wW2 = (const float*)d_in[5];
    const float* wB2 = (const float*)d_in[6];
    const float* F0  = (const float*)d_in[7];
    const float* Fb0 = (const float*)d_in[8];
    const float* F1  = (const float*)d_in[9];
    const float* Fb1 = (const float*)d_in[10];
    const float* F2  = (const float*)d_in[11];
    const float* Fb2 = (const float*)d_in[12];
    float* ws = (float*)d_ws;

    build_filters<<<1, 128, 0, stream>>>(wW0, wB0, wW1, wB1, wW2, wB2, ws);
    dim3 grid(NCHUNK, BATCH);
    fused_main<<<grid, NTHREADS, 0, stream>>>(n, F0, Fb0, F1, Fb1, F2, Fb2,
                                              ws, ws + WS_PART);
    finalize<<<1, 64, 0, stream>>>(ws + WS_PART, (float*)d_out);
}
